// ChebNet_4320737100467
// MI455X (gfx1250) — matrix operand order repaired and checked by reading, not compiled, not run on hardware
//
#include <hip/hip_runtime.h>

#define N_NODES 100000
#define F_IN    128
#define HID     32
#define NCLS    16
#define NTILES  (N_NODES / 16)   // 6250, exact

typedef __attribute__((ext_vector_type(16))) _Float16 v16h;
typedef __attribute__((ext_vector_type(8)))  float    v8f;

// ---------------------------------------------------------------- utilities

__global__ void cheb_zero(float* __restrict__ buf, int n) {
    int i = blockIdx.x * blockDim.x + threadIdx.x;
    if (i < n) buf[i] = 0.0f;
}

__global__ void cheb_deg_count(const int* __restrict__ row, float* __restrict__ deg, int E) {
    int e = blockIdx.x * blockDim.x + threadIdx.x;
    if (e < E) atomicAdd(&deg[row[e]], 1.0f);
}

__global__ void cheb_make_dis(float* __restrict__ deg_dis, int n) {
    int i = blockIdx.x * blockDim.x + threadIdx.x;
    if (i < n) {
        float d = deg_dis[i];
        deg_dis[i] = (d > 0.0f) ? rsqrtf(fmaxf(d, 1.0f)) : 0.0f;
    }
}

__global__ void cheb_relu(float* __restrict__ buf, int n) {
    int i = blockIdx.x * blockDim.x + threadIdx.x;
    if (i < n) buf[i] = fmaxf(buf[i], 0.0f);
}

// --------------------------------------------- weight pre-pack (f16, B layout)
// Layer A: [W0a | W1a] -> 16 tiles (nt 0..3, kt 0..3), per-lane v16h contiguous.
// B-tile 32x16 f16 (this code's K order, same as A): lane = N (both halves); lanes 0-15 K=0..15 (2/VGPR),
// lanes 16-31 K=16..31.

__global__ void cheb_pack_wA(const float* __restrict__ W0, const float* __restrict__ W1,
                             _Float16* __restrict__ out) {
    int t = blockIdx.x * blockDim.x + threadIdx.x;
    if (t >= 16 * 32) return;
    const int tile = t >> 5, lane = t & 31;
    const int nt = tile >> 2, kt = tile & 3;
    const float* W = (nt < 2) ? W0 : W1;
    const int m = lane & 15, hi = lane >> 4;
    const int col = ((nt & 1) << 4) + m;
    _Float16* dst = out + (size_t)t * 16;
#pragma unroll
    for (int v = 0; v < 8; ++v) {
        const int k = kt * 32 + (hi << 4) + (v << 1);
        dst[2 * v]     = (_Float16)W[(size_t)k * HID + col];
        dst[2 * v + 1] = (_Float16)W[(size_t)(k + 1) * HID + col];
    }
}

// Layer B: [W0b | W1b] (32x16 each) -> 2 tiles (nt 0..1).
__global__ void cheb_pack_wB(const float* __restrict__ W0, const float* __restrict__ W1,
                             _Float16* __restrict__ out) {
    int t = blockIdx.x * blockDim.x + threadIdx.x;
    if (t >= 2 * 32) return;
    const int nt = t >> 5, lane = t & 31;
    const float* W = nt ? W1 : W0;
    const int m = lane & 15, hi = lane >> 4;
    _Float16* dst = out + (size_t)t * 16;
#pragma unroll
    for (int v = 0; v < 8; ++v) {
        const int k = (hi << 4) + (v << 1);
        dst[2 * v]     = (_Float16)W[(size_t)k * NCLS + m];
        dst[2 * v + 1] = (_Float16)W[(size_t)(k + 1) * NCLS + m];
    }
}

// ------------------------------------------------------- layer A GEMM (WMMA)
// z = x @ W0a + ba   (N x 32),   p = x @ W1a   (N x 32)
// One wave per 16-row tile; K = 128 = 4 WMMA steps; 4 n-tiles of 16 cols.

__global__ void __launch_bounds__(256)
cheb_gemmA(const float* __restrict__ x, const v16h* __restrict__ pw,
           const float* __restrict__ ba,
           float* __restrict__ z, float* __restrict__ p)
{
    const int wave = (blockIdx.x * blockDim.x + threadIdx.x) >> 5;
    const int lane = threadIdx.x & 31;
    if (wave >= NTILES) return;            // wave-uniform: EXEC stays all-ones
    const int m  = lane & 15;
    const int hi = lane >> 4;

    const float* xrow = x + (size_t)(wave * 16 + m) * F_IN;
    if (wave + 8 < NTILES)                 // hint next tile (global_prefetch_b8)
        __builtin_prefetch(x + (size_t)((wave + 8) * 16 + m) * F_IN, 0, 1);

    // A tiles: 16x32 f16, lane = M (both halves), pair-packed K, k = 16*hi + s (same order as B pack)
    v16h A[4];
#pragma unroll
    for (int kt = 0; kt < 4; ++kt) {
#pragma unroll
        for (int v = 0; v < 8; ++v) {
            const int k = kt * 32 + (hi << 4) + (v << 1);
            A[kt][2 * v]     = (_Float16)xrow[k];
            A[kt][2 * v + 1] = (_Float16)xrow[k + 1];
        }
    }

#pragma unroll
    for (int nt = 0; nt < 4; ++nt) {
        const int col = ((nt & 1) << 4) + m;        // 0..31 within W0 or W1
        v8f acc = {};
#pragma unroll
        for (int kt = 0; kt < 4; ++kt) {
            const v16h B = pw[(size_t)((nt << 2) + kt) * 32 + lane]; // 2x b128
            acc = __builtin_amdgcn_wmma_f32_16x16x32_f16(
                false, A[kt], false, B, (short)0, acc, false, false);
        }
        // C/D layout: VGPR v -> M = hi*8 + v, lane -> N
        if (nt < 2) {
            const float bias = ba[col];
#pragma unroll
            for (int v = 0; v < 8; ++v)
                z[(size_t)(wave * 16 + (hi << 3) + v) * HID + col] = acc[v] + bias;
        } else {
#pragma unroll
            for (int v = 0; v < 8; ++v)
                p[(size_t)(wave * 16 + (hi << 3) + v) * HID + col] = acc[v];
        }
    }
}

// ------------------------------------------------------- layer B GEMM (WMMA)
// zb = h @ W0b + bb  (N x 16),   pb = h @ W1b  (N x 16);  K = 32, one step.

__global__ void __launch_bounds__(256)
cheb_gemmB(const float* __restrict__ h, const v16h* __restrict__ pw,
           const float* __restrict__ bb,
           float* __restrict__ zb, float* __restrict__ pb)
{
    const int wave = (blockIdx.x * blockDim.x + threadIdx.x) >> 5;
    const int lane = threadIdx.x & 31;
    if (wave >= NTILES) return;
    const int m  = lane & 15;
    const int hi = lane >> 4;

    const float* hrow = h + (size_t)(wave * 16 + m) * HID;
    v16h A;
#pragma unroll
    for (int v = 0; v < 8; ++v) {
        const int k = (hi << 4) + (v << 1);
        A[2 * v]     = (_Float16)hrow[k];
        A[2 * v + 1] = (_Float16)hrow[k + 1];
    }

#pragma unroll
    for (int nt = 0; nt < 2; ++nt) {
        const v16h B = pw[(size_t)nt * 32 + lane];
        v8f acc = {};
        acc = __builtin_amdgcn_wmma_f32_16x16x32_f16(
            false, A, false, B, (short)0, acc, false, false);
        if (nt == 0) {
            const float bias = bb[m];
#pragma unroll
            for (int v = 0; v < 8; ++v)
                zb[(size_t)(wave * 16 + (hi << 3) + v) * NCLS + m] = acc[v] + bias;
        } else {
#pragma unroll
            for (int v = 0; v < 8; ++v)
                pb[(size_t)(wave * 16 + (hi << 3) + v) * NCLS + m] = acc[v];
        }
    }
}

// -------------------------------------------------- edge gather/scatter-add
// z[row] += w * p[col], w = -dis[row]*dis[col]; F/4 threads per edge, float4.

template <int F>
__global__ void __launch_bounds__(256)
cheb_edge_scatter(const int* __restrict__ row, const int* __restrict__ col,
                  const float* __restrict__ dis, const float* __restrict__ p,
                  float* __restrict__ z, int E)
{
    const int cpe = F / 4;
    long t = (long)blockIdx.x * blockDim.x + threadIdx.x;
    if (t >= (long)E * cpe) return;
    const int e  = (int)(t / cpe);
    const int c  = (int)(t % cpe);
    const int r  = row[e];
    const int cc = col[e];
    const float w = -dis[r] * dis[cc];
    const float4 v = *reinterpret_cast<const float4*>(p + (size_t)cc * F + c * 4);
    float* dst = z + (size_t)r * F + c * 4;
    atomicAdd(dst + 0, w * v.x);
    atomicAdd(dst + 1, w * v.y);
    atomicAdd(dst + 2, w * v.z);
    atomicAdd(dst + 3, w * v.w);
}

// --------------------------------------------------------------- logsoftmax

__global__ void cheb_logsoftmax(const float* __restrict__ zb, float* __restrict__ out)
{
    int i = blockIdx.x * blockDim.x + threadIdx.x;
    if (i >= N_NODES) return;
    const float* r = zb + (size_t)i * NCLS;
    float mx = r[0];
#pragma unroll
    for (int j = 1; j < NCLS; ++j) mx = fmaxf(mx, r[j]);
    float s = 0.0f;
#pragma unroll
    for (int j = 0; j < NCLS; ++j) s += __expf(r[j] - mx);
    const float ls = __logf(s);
#pragma unroll
    for (int j = 0; j < NCLS; ++j)
        out[(size_t)i * NCLS + j] = r[j] - mx - ls;
}

// ------------------------------------------------------------------- driver

extern "C" void kernel_launch(void* const* d_in, const int* in_sizes, int n_in,
                              void* d_out, int out_size, void* d_ws, size_t ws_size,
                              hipStream_t stream)
{
    const float* x   = (const float*)d_in[0];
    const int*   ei  = (const int*)  d_in[1];
    const float* W0a = (const float*)d_in[2];
    const float* W1a = (const float*)d_in[3];
    const float* ba  = (const float*)d_in[4];
    const float* W0b = (const float*)d_in[5];
    const float* W1b = (const float*)d_in[6];
    const float* bb  = (const float*)d_in[7];

    const int E = in_sizes[1] / 2;
    const int* row = ei;
    const int* col = ei + E;

    float* ws  = (float*)d_ws;
    float* dis = ws;                              // N   (deg, then dis in place)
    float* z   = dis + N_NODES;                   // N*HID
    float* p   = z  + (size_t)N_NODES * HID;      // N*HID
    float* zb  = p  + (size_t)N_NODES * HID;      // N*NCLS
    float* pb  = zb + (size_t)N_NODES * NCLS;     // N*NCLS
    _Float16* pwA = (_Float16*)(pb + (size_t)N_NODES * NCLS); // 16*32*16 halves (32B-aligned)
    _Float16* pwB = pwA + 16 * 32 * 16;                       // 2*32*16 halves

    const int B = 256;
    const int gemm_blocks = (NTILES * 32 + B - 1) / B;

    // degree -> dis ; weight pre-pack into WMMA B layout (f16)
    cheb_zero<<<(N_NODES + B - 1) / B, B, 0, stream>>>(dis, N_NODES);
    cheb_deg_count<<<(E + B - 1) / B, B, 0, stream>>>(row, dis, E);
    cheb_make_dis<<<(N_NODES + B - 1) / B, B, 0, stream>>>(dis, N_NODES);
    cheb_pack_wA<<<2, B, 0, stream>>>(W0a, W1a, pwA);
    cheb_pack_wB<<<1, 64, 0, stream>>>(W0b, W1b, pwB);

    // layer A: z = x@W0a + ba ; p = x@W1a ; z += scatter(w * p[col]) ; relu
    cheb_gemmA<<<gemm_blocks, B, 0, stream>>>(x, (const v16h*)pwA, ba, z, p);
    {
        long th = (long)E * (HID / 4);
        cheb_edge_scatter<HID><<<(int)((th + B - 1) / B), B, 0, stream>>>(row, col, dis, p, z, E);
    }
    cheb_relu<<<(N_NODES * HID + B - 1) / B, B, 0, stream>>>(z, N_NODES * HID);

    // layer B: zb = h@W0b + bb ; pb = h@W1b ; zb += scatter(w * pb[col])
    cheb_gemmB<<<gemm_blocks, B, 0, stream>>>(z, (const v16h*)pwB, bb, zb, pb);
    {
        long th = (long)E * (NCLS / 4);
        cheb_edge_scatter<NCLS><<<(int)((th + B - 1) / B), B, 0, stream>>>(row, col, dis, pb, zb, E);
    }

    // log_softmax -> out
    cheb_logsoftmax<<<(N_NODES + B - 1) / B, B, 0, stream>>>(zb, (float*)d_out);
}
